// EGNN_50861002719986
// MI455X (gfx1250) — matrix operand order repaired and checked by reading, not compiled, not run on hardware
//
#include <hip/hip_runtime.h>
#include <hip/hip_bf16.h>
#include <math.h>

// ---------------------------------------------------------------------------
// EGNN on MI455X (gfx1250, wave32).  Dense GEMMs via v_wmma_f32_16x16x32_bf16,
// edge message passing via L2-resident f32 atomics.
// ---------------------------------------------------------------------------

typedef __attribute__((ext_vector_type(16))) __bf16 v16bf;
typedef __attribute__((ext_vector_type(8)))  float  v8f;

#define N_NODES 50000
#define N_EDGES 800000
#define MTILES  (N_NODES / 16)   // 3125 exactly

static inline size_t align256(size_t x) { return (x + 255) & ~(size_t)255; }

// ------------------------------ utility ------------------------------------
__global__ void egnn_fill0(float* __restrict__ p, size_t n) {
    size_t i = (size_t)blockIdx.x * blockDim.x + threadIdx.x;
    if (i < n) p[i] = 0.0f;
}

__global__ void egnn_count(const int* __restrict__ ei, float* __restrict__ cnt, int E) {
    int e = blockIdx.x * blockDim.x + threadIdx.x;
    if (e < E) atomicAdd(&cnt[ei[E + e]], 1.0f);
}

// ---------------------- pack A operand into WMMA frags ---------------------
// CDNA5 16-bit A 16x32 layout (ISA 7.12.2):
//   lanes 0-15  (m = lane):    elems 0-7 -> K 0-7,  elems 8-15 -> K 16-23
//   lanes 16-31 (m = lane-16): elems 0-7 -> K 8-15, elems 8-15 -> K 24-31
// One thread emits one lane-fragment (16 bf16 = 32B, contiguous).
// If cnt != nullptr, rows are scaled by 1/max(cnt[m],1)  (segment mean).
__global__ void egnn_pack_a(const float* __restrict__ src, const float* __restrict__ cnt,
                            __bf16* __restrict__ dst, int K, int KT) {
    int idx = blockIdx.x * blockDim.x + threadIdx.x;   // (mtile*KT + kt)*32 + lane
    int total = MTILES * KT * 32;
    if (idx >= total) return;
    int lane  = idx & 31;
    int kt    = (idx >> 5) % KT;
    int mtile = idx / (KT * 32);
    int half  = lane >> 4;
    int m     = mtile * 16 + (lane & 15);
    float s = 1.0f;
    if (cnt) s = 1.0f / fmaxf(cnt[m], 1.0f);
    const float* row = src + (size_t)m * K + kt * 32;
    __bf16* o = dst + (size_t)idx * 16;
#pragma unroll
    for (int i = 0; i < 16; ++i) {
        int k = (i < 8 ? i : i + 8) + half * 8;        // K(lane,i) per ISA layout
        o[i] = (__bf16)(row[k] * s);
    }
}

// ---------------------- pack B operand (weights, tiny) ---------------------
// 16-bit B 32x16: same K order as A: elems 0-7 -> K 8h+i, elems 8-15 -> K 16+8h+(i-8).
__global__ void egnn_pack_b(const float* __restrict__ w, __bf16* __restrict__ dst, int KT) {
    int idx = blockIdx.x * blockDim.x + threadIdx.x;   // (kt*4 + nt)*32 + lane
    int total = KT * 4 * 32;
    if (idx >= total) return;
    int lane = idx & 31;
    int nt   = (idx >> 5) & 3;
    int kt   = idx / 128;
    int half = lane >> 4;
    int n    = nt * 16 + (lane & 15);
    __bf16* o = dst + (size_t)idx * 16;
#pragma unroll
    for (int i = 0; i < 16; ++i) {
        int k = kt * 32 + (i < 8 ? i : i + 8) + half * 8;
        o[i] = (__bf16)w[(size_t)k * 64 + n];
    }
}

// --------------------------- WMMA GEMM + epilogue --------------------------
// out[M,64] = (A[M,K] @ B[K,64]) + bias, optional ReLU.  One 16-row tile per
// wave, 4 N-tiles of 16, K-loop in steps of 32 (KT iterations).
__global__ void egnn_gemm_wmma(const __bf16* __restrict__ pA,
                               const __bf16* __restrict__ pB,
                               const float* __restrict__ bias,
                               float* __restrict__ out,
                               int KT, int relu) {
    int wave  = threadIdx.x >> 5;
    int lane  = threadIdx.x & 31;
    int mtile = blockIdx.x * 4 + wave;
    if (mtile >= MTILES) return;                       // wave-uniform branch

    v8f acc0 = {}, acc1 = {}, acc2 = {}, acc3 = {};
    const __bf16* aBase = pA + (size_t)mtile * KT * 512 + (size_t)lane * 16;

    for (int kt = 0; kt < KT; ++kt) {
        if (kt + 1 < KT) __builtin_prefetch(aBase + (size_t)(kt + 1) * 512, 0, 1);
        v16bf a = *(const v16bf*)(aBase + (size_t)kt * 512);
        const __bf16* bBase = pB + (size_t)kt * 4 * 512 + (size_t)lane * 16;
        v16bf b0 = *(const v16bf*)(bBase + 0 * 512);
        v16bf b1 = *(const v16bf*)(bBase + 1 * 512);
        v16bf b2 = *(const v16bf*)(bBase + 2 * 512);
        v16bf b3 = *(const v16bf*)(bBase + 3 * 512);
        acc0 = __builtin_amdgcn_wmma_f32_16x16x32_bf16(false, a, false, b0, (short)0, acc0, false, false);
        acc1 = __builtin_amdgcn_wmma_f32_16x16x32_bf16(false, a, false, b1, (short)0, acc1, false, false);
        acc2 = __builtin_amdgcn_wmma_f32_16x16x32_bf16(false, a, false, b2, (short)0, acc2, false, false);
        acc3 = __builtin_amdgcn_wmma_f32_16x16x32_bf16(false, a, false, b3, (short)0, acc3, false, false);
    }

    // C/D layout: lane (l&15) = column, VGPR r -> row r + 8*(l>>4)
    int n0   = lane & 15;
    int half = lane >> 4;
    v8f accs[4] = {acc0, acc1, acc2, acc3};
#pragma unroll
    for (int nt = 0; nt < 4; ++nt) {
        int n = nt * 16 + n0;
        float bv = bias[n];
#pragma unroll
        for (int r = 0; r < 8; ++r) {
            int m = mtile * 16 + r + 8 * half;
            float v = accs[nt][r] + bv;
            if (relu) v = fmaxf(v, 0.0f);
            out[(size_t)m * 64 + n] = v;
        }
    }
}

// ------------------------- edge message scatter ----------------------------
// msg[e] = edge_attr[e, c] * h[src[e], f]  scattered-summed into agg[tgt, c*64+f].
// 16 threads per edge, float4 per thread: coalesced 256B row gather,
// 12 global_atomic_add_f32 per thread into L2-resident agg.
__global__ void egnn_scatter(const float* __restrict__ h, const int* __restrict__ ei,
                             const float* __restrict__ ea, float* __restrict__ agg, int E) {
    long long t = (long long)blockIdx.x * blockDim.x + threadIdx.x;
    int e = (int)(t >> 4);
    if (e >= E) return;
    int f0  = ((int)t & 15) * 4;
    int src = ei[e];
    int tgt = ei[E + e];
    float a0 = ea[(size_t)e * 3 + 0];
    float a1 = ea[(size_t)e * 3 + 1];
    float a2 = ea[(size_t)e * 3 + 2];
    const float4 v = *(const float4*)(h + (size_t)src * 64 + f0);
    float* base = agg + (size_t)tgt * 192 + f0;
    const float xs[4] = {v.x, v.y, v.z, v.w};
#pragma unroll
    for (int j = 0; j < 4; ++j) {
        atomicAdd(base + j,        a0 * xs[j]);
        atomicAdd(base + 64 + j,   a1 * xs[j]);
        atomicAdd(base + 128 + j,  a2 * xs[j]);
    }
}

// ------------------------------ L2 normalize -------------------------------
// One wave32 per node row of 64; shuffle reduction for sum of squares.
__global__ void egnn_l2norm(float* __restrict__ h, int Nn) {
    int w    = (int)(((long long)blockIdx.x * blockDim.x + threadIdx.x) >> 5);
    int lane = threadIdx.x & 31;
    if (w >= Nn) return;                               // wave-uniform
    float* row = h + (size_t)w * 64;
    float x0 = row[lane];
    float x1 = row[lane + 32];
    float ss = x0 * x0 + x1 * x1;
#pragma unroll
    for (int off = 16; off > 0; off >>= 1) ss += __shfl_xor(ss, off, 32);
    float scale = 1.0f / fmaxf(sqrtf(ss), 1e-12f);
    row[lane]      = x0 * scale;
    row[lane + 32] = x1 * scale;
}

// ------------------------------- launcher ----------------------------------
extern "C" void kernel_launch(void* const* d_in, const int* in_sizes, int n_in,
                              void* d_out, int out_size, void* d_ws, size_t ws_size,
                              hipStream_t stream) {
    (void)in_sizes; (void)n_in; (void)out_size; (void)ws_size;
    const float* x     = (const float*)d_in[0];
    const int*   ei    = (const int*)d_in[1];
    const float* ea    = (const float*)d_in[2];
    const float* pre_w = (const float*)d_in[3];
    const float* pre_b = (const float*)d_in[4];
    const float* w1 = (const float*)d_in[5];  const float* b1 = (const float*)d_in[6];
    const float* w2 = (const float*)d_in[7];  const float* b2 = (const float*)d_in[8];
    const float* w3 = (const float*)d_in[9];  const float* b3 = (const float*)d_in[10];
    float* out = (float*)d_out;

    const int Nn = N_NODES, E = N_EDGES;

    // ---- workspace carve-up (all offsets 256B aligned) ----
    char* ws = (char*)d_ws;
    float* cnt = (float*)ws;                     ws += align256((size_t)Nn * sizeof(float));
    float* h   = (float*)ws;                     ws += align256((size_t)Nn * 64 * sizeof(float));
    float* agg = (float*)ws;                     ws += align256((size_t)Nn * 192 * sizeof(float));
    __bf16* pA = (__bf16*)ws;                    ws += align256((size_t)MTILES * 6 * 512 * sizeof(__bf16));
    __bf16* pB = (__bf16*)ws;                    /* 6*4*512 bf16 = 24 KB */

    const int TB = 256;
    // ---- edge counts (shared by all three layers) ----
    egnn_fill0<<<(Nn + TB - 1) / TB, TB, 0, stream>>>(cnt, (size_t)Nn);
    egnn_count<<<(E + TB - 1) / TB, TB, 0, stream>>>(ei, cnt, E);

    // ---- linear_pre: h = x @ pre_w + pre_b  (K=128 -> KT=4, no relu) ----
    {
        int KT = 4;
        egnn_pack_b<<<(KT * 128 + TB - 1) / TB, TB, 0, stream>>>(pre_w, pB, KT);
        egnn_pack_a<<<(MTILES * KT * 32 + TB - 1) / TB, TB, 0, stream>>>(x, nullptr, pA, 128, KT);
        egnn_gemm_wmma<<<(MTILES + 3) / 4, 128, 0, stream>>>(pA, pB, pre_b, h, KT, 0);
    }

    // ---- three SAGE conv layers (K=192 -> KT=6) ----
    const float* Ws[3] = {w1, w2, w3};
    const float* Bs[3] = {b1, b2, b3};
    for (int layer = 0; layer < 3; ++layer) {
        int relu = (layer < 2) ? 1 : 0;
        float* dst = (layer < 2) ? h : out;
        int KT = 6;
        egnn_fill0<<<((size_t)Nn * 192 + TB - 1) / TB, TB, 0, stream>>>(agg, (size_t)Nn * 192);
        egnn_scatter<<<(int)(((long long)E * 16 + TB - 1) / TB), TB, 0, stream>>>(h, ei, ea, agg, E);
        egnn_pack_a<<<(MTILES * KT * 32 + TB - 1) / TB, TB, 0, stream>>>(agg, cnt, pA, 192, KT);
        egnn_pack_b<<<(KT * 128 + TB - 1) / TB, TB, 0, stream>>>(Ws[layer], pB, KT);
        egnn_gemm_wmma<<<(MTILES + 3) / 4, 128, 0, stream>>>(pA, pB, Bs[layer], dst, KT, relu);
    }

    // ---- row-wise L2 normalize (in place on d_out) ----
    egnn_l2norm<<<(Nn * 32 + TB - 1) / TB, TB, 0, stream>>>(out, Nn);
}
